// SinkhornLoss_84954453114886
// MI455X (gfx1250) — compile-verified, matrix operand order checked by reading, not run on hardware
//
#include <hip/hip_runtime.h>
#include <math.h>

typedef __attribute__((ext_vector_type(16))) _Float16 v16h;
typedef __attribute__((ext_vector_type(8)))  _Float16 v8h;
typedef __attribute__((ext_vector_type(8)))  float    v8f;

#define NN 4096   // rows of O == rows of T == both matvec dims
#define DD 512    // feature dim (GEMM K)

// ---------------- row L2-normalize + cast to f16 (fold 1/||row|| into data) ----
__global__ void sk_nrm_kernel(const float* __restrict__ src, _Float16* __restrict__ dst) {
    __shared__ float red[128];
    const int row = blockIdx.x;
    const int tid = threadIdx.x;               // 128 threads
    const float* r = src + (size_t)row * DD;
    float v[4];
    float s = 0.f;
#pragma unroll
    for (int k = 0; k < 4; ++k) { v[k] = r[tid + 128 * k]; s += v[k] * v[k]; }
    red[tid] = s;
    __syncthreads();
    for (int off = 64; off > 0; off >>= 1) {
        if (tid < off) red[tid] += red[tid + off];
        __syncthreads();
    }
    const float rn = 1.0f / fmaxf(sqrtf(red[0]), 1e-12f);
    _Float16* w = dst + (size_t)row * DD;
#pragma unroll
    for (int k = 0; k < 4; ++k) w[tid + 128 * k] = (_Float16)(v[k] * rn);
}

// ---------------- fragment helpers (ISA 7.12.2 16-bit A/B layout) -------------
static __device__ inline v16h sk_combine(v8h lo, v8h hi) {
    return __builtin_shufflevector(lo, hi, 0, 1, 2, 3, 4, 5, 6, 7,
                                          8, 9, 10, 11, 12, 13, 14, 15);
}
static __device__ inline v16h sk_load_frag(const _Float16* p) {
    // lane already points at (row, k0 + laneHi*8); +16 halves = K+16 block
    v8h lo = *(const v8h*)(p);
    v8h hi = *(const v8h*)(p + 16);
    return sk_combine(lo, hi);
}

// ---------------- GEMM (O * T^T) + exp epilogue:  K = exp(2*cos - 2) ----------
// block = 256 threads = 8 waves arranged 2 (rows) x 4 (cols);
// wave tile = 16 rows x 64 cols (1 A frag reused over 4 B frags -> 4 wmma/step)
__global__ void __launch_bounds__(256)
sk_gemm_exp_kernel(const _Float16* __restrict__ A, const _Float16* __restrict__ B,
                   float* __restrict__ Kmat) {
    const int tid    = threadIdx.x;
    const int wave   = tid >> 5;
    const int lane   = tid & 31;
    const int laneLo = lane & 15;
    const int laneHi = lane >> 4;
    const int rowBase = blockIdx.y * 32  + (wave & 1) * 16;
    const int colBase = blockIdx.x * 256 + (wave >> 1) * 64;

    v8f c0 = {}, c1 = {}, c2 = {}, c3 = {};

    const _Float16* aP  = A + (size_t)(rowBase + laneLo) * DD + laneHi * 8;
    const _Float16* b0P = B + (size_t)(colBase +  0 + laneLo) * DD + laneHi * 8;
    const _Float16* b1P = B + (size_t)(colBase + 16 + laneLo) * DD + laneHi * 8;
    const _Float16* b2P = B + (size_t)(colBase + 32 + laneLo) * DD + laneHi * 8;
    const _Float16* b3P = B + (size_t)(colBase + 48 + laneLo) * DD + laneHi * 8;

    for (int k0 = 0; k0 < DD; k0 += 32) {
        v16h a  = sk_load_frag(aP  + k0);
        v16h b0 = sk_load_frag(b0P + k0);
        v16h b1 = sk_load_frag(b1P + k0);
        v16h b2 = sk_load_frag(b2P + k0);
        v16h b3 = sk_load_frag(b3P + k0);
        c0 = __builtin_amdgcn_wmma_f32_16x16x32_f16(false, a, false, b0, (short)0, c0, false, false);
        c1 = __builtin_amdgcn_wmma_f32_16x16x32_f16(false, a, false, b1, (short)0, c1, false, false);
        c2 = __builtin_amdgcn_wmma_f32_16x16x32_f16(false, a, false, b2, (short)0, c2, false, false);
        c3 = __builtin_amdgcn_wmma_f32_16x16x32_f16(false, a, false, b3, (short)0, c3, false, false);
    }

    // C/D layout: VGPR r -> (M = rowBase + laneHi*8 + r, N = colBase + tile*16 + laneLo)
#pragma unroll
    for (int r = 0; r < 8; ++r) {
        const size_t m = (size_t)(rowBase + laneHi * 8 + r) * NN;
        const int    c = colBase + laneLo;
        Kmat[m + c +  0] = __expf(2.0f * c0[r] - 2.0f);
        Kmat[m + c + 16] = __expf(2.0f * c1[r] - 2.0f);
        Kmat[m + c + 32] = __expf(2.0f * c2[r] - 2.0f);
        Kmat[m + c + 48] = __expf(2.0f * c3[r] - 2.0f);
    }
}

// ---------------- Sinkhorn pieces --------------------------------------------
__global__ void sk_init_left(float* __restrict__ left) {
    left[blockIdx.x * 256 + threadIdx.x] = 1.0f;
}

// partial column sums: s_j = sum_i left_i K_ij over a 128-row chunk (coalesced)
__global__ void sk_colpart_kernel(const float* __restrict__ Kmat,
                                  const float* __restrict__ left,
                                  float* __restrict__ part) {
    const int col = blockIdx.x * 256 + threadIdx.x;
    const int r0  = blockIdx.y * 128;
    float acc = 0.f;
    for (int r = 0; r < 128; ++r)
        acc = fmaf(left[r0 + r], Kmat[(size_t)(r0 + r) * NN + col], acc);
    part[(size_t)blockIdx.y * NN + col] = acc;
}

// right_j = (1/M) / sum_c part[c][j]    (deterministic fixed-order sum)
__global__ void sk_colfin_kernel(const float* __restrict__ part,
                                 float* __restrict__ right) {
    const int col = blockIdx.x * 256 + threadIdx.x;
    float s = 0.f;
#pragma unroll
    for (int c = 0; c < 32; ++c) s += part[(size_t)c * NN + col];
    right[col] = (1.0f / (float)NN) / s;
}

// left_i = (1/N) / (K_i . right)   -- one block per row, LDS tree reduce
__global__ void sk_rowmv_kernel(const float* __restrict__ Kmat,
                                const float* __restrict__ right,
                                float* __restrict__ left) {
    __shared__ float red[256];
    const int row = blockIdx.x, tid = threadIdx.x;
    const float* kr = Kmat + (size_t)row * NN;
    float acc = 0.f;
#pragma unroll
    for (int j = 0; j < NN; j += 256)
        acc = fmaf(kr[j + tid], right[j + tid], acc);
    red[tid] = acc; __syncthreads();
    for (int off = 128; off > 0; off >>= 1) {
        if (tid < off) red[tid] += red[tid + off];
        __syncthreads();
    }
    if (tid == 0) left[row] = (1.0f / (float)NN) / red[0];
}

// ---------------- loss = sum_ij (-log K) * K * left_i * right_j ---------------
__global__ void sk_losspart_kernel(const float* __restrict__ Kmat,
                                   const float* __restrict__ left,
                                   const float* __restrict__ right,
                                   float* __restrict__ lp) {
    __shared__ float red[256];
    const int row = blockIdx.x, tid = threadIdx.x;
    const float* kr = Kmat + (size_t)row * NN;
    float acc = 0.f;
#pragma unroll
    for (int j = 0; j < NN; j += 256) {
        const float k = kr[j + tid];
        acc = fmaf((-__logf(k)) * k, right[j + tid], acc);   // C = -log(K)
    }
    red[tid] = acc; __syncthreads();
    for (int off = 128; off > 0; off >>= 1) {
        if (tid < off) red[tid] += red[tid + off];
        __syncthreads();
    }
    if (tid == 0) lp[row] = left[row] * red[0];
}

__global__ void sk_lossfin_kernel(const float* __restrict__ lp, float* __restrict__ out) {
    __shared__ float red[256];
    const int tid = threadIdx.x;
    float acc = 0.f;
#pragma unroll
    for (int i = 0; i < 16; ++i) acc += lp[tid + 256 * i];
    red[tid] = acc; __syncthreads();
    for (int off = 128; off > 0; off >>= 1) {
        if (tid < off) red[tid] += red[tid + off];
        __syncthreads();
    }
    if (tid == 0) out[0] = red[0];
}

// ---------------- host launcher ----------------------------------------------
extern "C" void kernel_launch(void* const* d_in, const int* in_sizes, int n_in,
                              void* d_out, int out_size, void* d_ws, size_t ws_size,
                              hipStream_t stream) {
    (void)in_sizes; (void)n_in; (void)out_size; (void)ws_size;
    const float* O = (const float*)d_in[0];
    const float* T = (const float*)d_in[1];

    char* ws = (char*)d_ws;
    _Float16* Oh   = (_Float16*)ws;                                   // 4 MB
    _Float16* Th   = (_Float16*)(ws + (size_t)NN * DD * 2);           // 4 MB
    float*    Kmat = (float*)   (ws + (size_t)2 * NN * DD * 2);       // 64 MB
    char* p = ws + (size_t)2 * NN * DD * 2 + (size_t)NN * NN * 4;
    float* part  = (float*)p;  p += (size_t)32 * NN * 4;              // 512 KB
    float* left  = (float*)p;  p += (size_t)NN * 4;
    float* right = (float*)p;  p += (size_t)NN * 4;
    float* lp    = (float*)p;

    sk_nrm_kernel<<<NN, 128, 0, stream>>>(O, Oh);
    sk_nrm_kernel<<<NN, 128, 0, stream>>>(T, Th);

    sk_gemm_exp_kernel<<<dim3(NN / 256, NN / 32), 256, 0, stream>>>(Oh, Th, Kmat);

    sk_init_left<<<NN / 256, 256, 0, stream>>>(left);
    for (int it = 0; it < 10; ++it) {
        sk_colpart_kernel<<<dim3(NN / 256, 32), 256, 0, stream>>>(Kmat, left, part);
        sk_colfin_kernel<<<NN / 256, 256, 0, stream>>>(part, right);
        sk_rowmv_kernel<<<NN, 256, 0, stream>>>(Kmat, right, left);
    }

    sk_losspart_kernel<<<NN, 256, 0, stream>>>(Kmat, left, right, lp);
    sk_lossfin_kernel<<<1, 256, 0, stream>>>(lp, (float*)d_out);
}
